// VectorQuantizerEMA_88175678587408
// MI455X (gfx1250) — matrix operand order repaired and checked by reading, not compiled, not run on hardware
//
#include <hip/hip_runtime.h>
#include <hip/hip_bf16.h>

// ---------------------------------------------------------------------------
// VectorQuantizerEMA for MI455X (gfx1250, wave32, WMMA)
// N=65536 inputs, K=2048 codes, D=128. Distance GEMM via v_wmma_f32_16x16x32_bf16.
// Each wave register-blocks 4 row-tiles (64 rows) to reuse codebook A-fragments
// 4x, cutting L2 codebook traffic from 2 GB to 512 MB.
// ---------------------------------------------------------------------------

#define VQ_N 65536
#define VQ_K 2048
#define VQ_D 128
#define VQ_TILES (VQ_N / 16)      // 4096 row tiles of 16 rows
#define VQ_CHUNKS (VQ_K / 16)     // 128 code chunks of 16 codes
#define VQ_RT 4                   // row tiles per wave (register blocking)

typedef __attribute__((ext_vector_type(16))) __bf16 v16bf;
typedef __attribute__((ext_vector_type(8)))  float  v8f;

// -------------------- fragment load helpers --------------------
__device__ __forceinline__ v16bf frag_2x8(const __bf16* lo, const __bf16* hi) {
  union { float4 f[2]; v16bf v; } u;
  u.f[0] = *reinterpret_cast<const float4*>(lo);   // 8 bf16 (16 B)
  u.f[1] = *reinterpret_cast<const float4*>(hi);   // 8 bf16 (16 B)
  return u.v;
}
__device__ __forceinline__ v16bf frag_16(const __bf16* p) {
  union { float4 f[2]; v16bf v; } u;
  const float4* q = reinterpret_cast<const float4*>(p);
  u.f[0] = q[0]; u.f[1] = q[1];                    // 16 bf16 (32 B) contiguous
  return u.v;
}

// -------------------- prep: embedding -> bf16, ||e||^2, zero accumulators ---
__global__ __launch_bounds__(128)
void k_prep_embed(const float* __restrict__ emb, __bf16* __restrict__ ebf,
                  float* __restrict__ enorm, float* __restrict__ dw,
                  float* __restrict__ dw_embed, float* __restrict__ scal) {
  int k = blockIdx.x, d = threadIdx.x;
  float e = emb[k * VQ_D + d];
  ebf[k * VQ_D + d] = (__bf16)e;
  dw_embed[k * VQ_D + d] = 0.0f;
  if (d == 0) dw[k] = 0.0f;
  if (k == 0 && d < 4) scal[d] = 0.0f;   // [0]=loss_sum, [1]=n_sum
  __shared__ float s[128];
  s[d] = e * e;
  __syncthreads();
  for (int off = 64; off > 0; off >>= 1) {
    if (d < off) s[d] += s[d + off];
    __syncthreads();
  }
  if (d == 0) enorm[k] = s[0];
}

// -------------------- prep: X -> bf16, tiled [tile][r][d] ---------
// LDS-staged so both the global read and the bf16 write are coalesced.
__global__ __launch_bounds__(256)
void k_prep_x(const float* __restrict__ x, __bf16* __restrict__ xT) {
  __shared__ float s[16][VQ_D + 1];   // +1 pad: stride 129 avoids bank conflicts
  int t = blockIdx.x;                 // one 16-row tile per block
  for (int i = threadIdx.x; i < 16 * VQ_D; i += 256) {
    int r = i >> 7, d = i & (VQ_D - 1);
    s[r][d] = x[(t * 16 + r) * VQ_D + d];       // coalesced read
  }
  __syncthreads();
  for (int i = threadIdx.x; i < 16 * VQ_D; i += 256) {
    int r = i >> 7, d = i & (VQ_D - 1);
    xT[t * (16 * VQ_D) + i] = (__bf16)s[r][d];  // coalesced write, LDS stride 129
  }
}

// -------------------- WMMA distance + argmin --------------------------------
// One wave32 per VQ_RT row tiles. A = codebook tile (16 codes x 32 d, bf16),
// B = xT tiles (32 d x 16 rows, bf16), C init = -0.5*||e||^2 so acc = -0.5*dist
// and argmin(dist) == argmax(acc).
__global__ __launch_bounds__(256)
void k_argmin(const __bf16* __restrict__ ebf, const __bf16* __restrict__ xT,
              const float* __restrict__ enorm, int* __restrict__ idx,
              float* __restrict__ out_idx_f) {
  const int lane = threadIdx.x & 31;
  const int wv = blockIdx.x * 8 + (threadIdx.x >> 5);
  const int tbase = wv * VQ_RT;
  const int hi16 = (lane & 16) ? 1 : 0;

  // B fragments (X tiles), hoisted: lane holds row (lane&15), d = hi16*8 + {0..7, 16..23} (+32c)
  const int dB = (lane & 15) * VQ_D + hi16 * 8;
  v16bf b[VQ_RT][4];
#pragma unroll
  for (int r = 0; r < VQ_RT; ++r) {
    const __bf16* xt = xT + (size_t)(tbase + r) * (16 * VQ_D);
#pragma unroll
    for (int c = 0; c < 4; ++c) b[r][c] = frag_2x8(xt + dB + 32 * c, xt + dB + 32 * c + 16);
  }

  const int cA = lane & 15;       // A: code row within chunk
  const int dA = hi16 * 8;        // A: K-offset split per lane half
  const int dC = hi16 * 8;        // C/D: VGPR v -> code row v + dC

  float best[VQ_RT];
  int bestIdx[VQ_RT];
#pragma unroll
  for (int r = 0; r < VQ_RT; ++r) { best[r] = -3.4e38f; bestIdx[r] = 0; }

  for (int chunk = 0; chunk < VQ_CHUNKS; ++chunk) {
    // C init: -0.5 * ||e_code||^2 for codes chunk*16 + dC + v (same for all r)
    const float4 e0 = *reinterpret_cast<const float4*>(enorm + chunk * 16 + dC);
    const float4 e1 = *reinterpret_cast<const float4*>(enorm + chunk * 16 + dC + 4);
    v8f cinit;
    cinit[0] = -0.5f * e0.x; cinit[1] = -0.5f * e0.y;
    cinit[2] = -0.5f * e0.z; cinit[3] = -0.5f * e0.w;
    cinit[4] = -0.5f * e1.x; cinit[5] = -0.5f * e1.y;
    cinit[6] = -0.5f * e1.z; cinit[7] = -0.5f * e1.w;

    v8f acc[VQ_RT];
#pragma unroll
    for (int r = 0; r < VQ_RT; ++r) acc[r] = cinit;

    const __bf16* arow = ebf + (chunk * 16 + cA) * VQ_D + dA;
#pragma unroll
    for (int c = 0; c < 4; ++c) {
      v16bf a = frag_2x8(arow + c * 32, arow + c * 32 + 16);  // shared A-frag
#pragma unroll
      for (int r = 0; r < VQ_RT; ++r) {
        acc[r] = __builtin_amdgcn_wmma_f32_16x16x32_bf16(
            /*neg_a=*/false, a, /*neg_b=*/false, b[r][c],
            /*c_mod=*/(short)0, acc[r], /*reuse_a=*/false, /*reuse_b=*/false);
      }
    }

    // acc[r][v] = -0.5*dist(code = chunk*16+dC+v, row = (tbase+r)*16+(lane&15))
#pragma unroll
    for (int r = 0; r < VQ_RT; ++r) {
#pragma unroll
      for (int v = 0; v < 8; ++v) {
        float val = acc[r][v];
        if (val > best[r]) { best[r] = val; bestIdx[r] = chunk * 16 + dC + v; }
      }
    }
  }

  // merge the two half-wave candidates (codes v+0..7 vs v+8..15 per chunk)
#pragma unroll
  for (int r = 0; r < VQ_RT; ++r) {
    float pv = __shfl_xor(best[r], 16, 32);
    int   pi = __shfl_xor(bestIdx[r], 16, 32);
    if (pv > best[r] || (pv == best[r] && pi < bestIdx[r])) {
      best[r] = pv; bestIdx[r] = pi;
    }
    if (lane < 16) {
      int row = (tbase + r) * 16 + lane;
      idx[row] = bestIdx[r];
      out_idx_f[row] = (float)bestIdx[r];
    }
  }
}

// -------------------- gather + loss + segment sums --------------------------
__global__ __launch_bounds__(128)
void k_gather(const float* __restrict__ x, const float* __restrict__ emb,
              const int* __restrict__ idx, float* __restrict__ out_q,
              float* __restrict__ dw, float* __restrict__ dw_embed,
              float* __restrict__ loss_sum) {
  int n = blockIdx.x, d = threadIdx.x;
  int k = idx[n];
  float q  = emb[k * VQ_D + d];
  float xv = x[n * VQ_D + d];
  out_q[n * VQ_D + d] = q;                 // quantized_st == quantized numerically
  atomicAdd(&dw_embed[k * VQ_D + d], xv);  // segment_sum(flat)
  if (d == 0) atomicAdd(&dw[k], 1.0f);     // segment_sum(ones)
  float diff = q - xv;
  __shared__ float s[128];
  s[d] = diff * diff;
  __syncthreads();
  for (int off = 64; off > 0; off >>= 1) {
    if (d < off) s[d] += s[d + off];
    __syncthreads();
  }
  if (d == 0) atomicAdd(loss_sum, s[0]);
}

// -------------------- EMA cluster size + total n ----------------------------
__global__ __launch_bounds__(256)
void k_ema_cs(const float* __restrict__ cs, const float* __restrict__ dw,
              float* __restrict__ out_ncs, float* __restrict__ scal) {
  int k = blockIdx.x * 256 + threadIdx.x;
  float v = cs[k] * 0.99f + 0.01f * dw[k];
  out_ncs[k] = v;
  __shared__ float s[256];
  s[threadIdx.x] = v;
  __syncthreads();
  for (int off = 128; off > 0; off >>= 1) {
    if (threadIdx.x < off) s[threadIdx.x] += s[threadIdx.x + off];
    __syncthreads();
  }
  if (threadIdx.x == 0) atomicAdd(&scal[1], s[0]);  // n_sum
}

// -------------------- finalize: embed_avg EMA, normalize, loss --------------
__global__ __launch_bounds__(256)
void k_final(const float* __restrict__ ea, const float* __restrict__ dw_embed,
             const float* __restrict__ out_ncs, const float* __restrict__ scal,
             float* __restrict__ out_nea, float* __restrict__ out_ne,
             float* __restrict__ out_loss) {
  int i = blockIdx.x * 256 + threadIdx.x;
  int k = i >> 7;  // /VQ_D
  float nea = ea[i] * 0.99f + 0.01f * dw_embed[i];
  out_nea[i] = nea;
  float nsum = scal[1];
  float sm = (out_ncs[k] + 1e-5f) / (nsum + (float)VQ_K * 1e-5f) * nsum;
  out_ne[i] = nea / sm;
  if (i == 0) *out_loss = 2.0f * scal[0] / ((float)VQ_N * (float)VQ_D);
}

// ---------------------------------------------------------------------------
extern "C" void kernel_launch(void* const* d_in, const int* in_sizes, int n_in,
                              void* d_out, int out_size, void* d_ws, size_t ws_size,
                              hipStream_t stream) {
  const float* x    = (const float*)d_in[0];  // [N,D]
  const float* emb  = (const float*)d_in[1];  // [K,D]
  const float* cs   = (const float*)d_in[2];  // [K]
  const float* ea   = (const float*)d_in[3];  // [K,D]
  float* out = (float*)d_out;

  // output layout (floats), concatenated in reference return order
  const size_t OQ   = 0;                         // quantized_st  [N*D]
  const size_t OL   = OQ + (size_t)VQ_N * VQ_D;  // loss          [1]
  const size_t OI   = OL + 1;                    // indices       [N]
  const size_t ONE  = OI + VQ_N;                 // new_embedding [K*D]
  const size_t ONCS = ONE + (size_t)VQ_K * VQ_D; // new_cluster   [K]
  const size_t ONEA = ONCS + VQ_K;               // new_embed_avg [K*D]

  // workspace layout (bytes)
  char* ws = (char*)d_ws;
  __bf16* ebf      = (__bf16*)(ws);                         // K*D*2 = 0.5 MB
  __bf16* xT       = (__bf16*)(ws + 524288);                // N*D*2 = 16 MB
  float*  enorm    = (float*)(ws + 17301504);               // K*4
  int*    idx      = (int*)  (ws + 17309696);               // N*4
  float*  dw       = (float*)(ws + 17571840);               // K*4
  float*  dw_embed = (float*)(ws + 17580032);               // K*D*4 = 1 MB
  float*  scal     = (float*)(ws + 18628608);               // [loss_sum, n_sum]

  k_prep_embed<<<VQ_K, 128, 0, stream>>>(emb, ebf, enorm, dw, dw_embed, scal);
  k_prep_x<<<VQ_TILES, 256, 0, stream>>>(x, xT);
  k_argmin<<<VQ_TILES / (VQ_RT * 8), 256, 0, stream>>>(ebf, xT, enorm, idx, out + OI);
  k_gather<<<VQ_N, 128, 0, stream>>>(x, emb, idx, out + OQ, dw, dw_embed, scal);
  k_ema_cs<<<VQ_K / 256, 256, 0, stream>>>(cs, dw, out + ONCS, scal);
  k_final<<<(VQ_K * VQ_D) / 256, 256, 0, stream>>>(ea, dw_embed, out + ONCS, scal,
                                                   out + ONEA, out + ONE, out + OL);
}
